// PyTorchRNNDecoder_76622216560968
// MI455X (gfx1250) — matrix operand order repaired and checked by reading, not compiled, not run on hardware
//
#include <hip/hip_runtime.h>
#include <hip/hip_bf16.h>

typedef __attribute__((ext_vector_type(16))) __bf16 v16bf;
typedef __attribute__((ext_vector_type(8)))  float  v8f;

constexpr int N_  = 16;
constexpr int Ti_ = 1000;
constexpr int To_ = 200;
constexpr int D_  = 512;
constexpr int H_  = 512;
constexpr int V_  = 8000;
constexpr int G4H = 4 * H_;          // 2048 gate width
constexpr int KX0 = D_ + H_;         // 1024 layer-0 input width

__device__ __forceinline__ unsigned short f2bf(float f) {
  unsigned int u = __float_as_uint(f);
  unsigned int r = u + 0x7FFFu + ((u >> 16) & 1u);   // round-to-nearest-even
  return (unsigned short)(r >> 16);
}

__device__ __forceinline__ float sigf(float x) {
  return 1.0f / (1.0f + __expf(-x));
}

__device__ __forceinline__ v8f wmma_bf16(v16bf a, v16bf b, v8f c) {
  return __builtin_amdgcn_wmma_f32_16x16x32_bf16(false, a, false, b,
                                                 (short)0, c, false, false);
}

// A fragment: 16x32 bf16, lane m = lane&15, kbase = (lane>>4)*8
// elements 0..7  -> K = k0 + kbase + i ; elements 8..15 -> K = k0 + 16 + kbase + (i-8)
__device__ __forceinline__ v16bf load_frag_a(const unsigned short* __restrict__ x,
                                             int ld, int k0, int lane) {
  int m  = lane & 15;
  int kb = (lane >> 4) << 3;
  const unsigned short* p = x + (size_t)m * ld + k0 + kb;
  union { v16bf v; uint4 q[2]; } u;
  u.q[0] = *reinterpret_cast<const uint4*>(p);
  u.q[1] = *reinterpret_cast<const uint4*>(p + 16);
  return u.v;
}

// B fragment: 32x16 bf16 from K-major (pre-transposed) weight Wt[K][Nout]:
// lane n = n0 + (lane&15), kbase = (lane>>4)*8; elements 0..7 -> K = k0 + kbase + i,
// elements 8..15 -> K = k0 + 16 + kbase + (i-8)
__device__ __forceinline__ v16bf load_frag_b(const unsigned short* __restrict__ wt,
                                             int ld, int k0, int n0, int lane) {
  int kb = (lane >> 4) << 3;
  const unsigned short* p = wt + (size_t)(k0 + kb) * ld + n0 + (lane & 15);
  union { v16bf v; unsigned short s[16]; } u;
#pragma unroll
  for (int i = 0; i < 8; ++i) {
    u.s[i]     = p[(size_t)i * ld];
    u.s[8 + i] = p[(size_t)(16 + i) * ld];
  }
  return u.v;
}

// Ping-pong pipelined K-loop: two fragment buffer sets, loads land directly in
// the set consumed two WMMA-groups later -> no register copies, no WAR NOPs.
// nT must be even and >= 2 (all call sites satisfy this).
template <int Q>
__device__ __forceinline__ void gemm_pipe(const unsigned short* __restrict__ asrc, int lda,
                                          const unsigned short* __restrict__ wt, int ldb,
                                          int nT, int colb, int colstride, int lane,
                                          v8f* acc) {
  v16bf a0, a1, b0[Q], b1[Q];
  a0 = load_frag_a(asrc, lda, 0, lane);
#pragma unroll
  for (int q = 0; q < Q; ++q) b0[q] = load_frag_b(wt, ldb, 0, q * colstride + colb, lane);

  int ti = 0;
  for (; ti < nT - 2; ti += 2) {
    a1 = load_frag_a(asrc, lda, (ti + 1) << 5, lane);
#pragma unroll
    for (int q = 0; q < Q; ++q)
      b1[q] = load_frag_b(wt, ldb, (ti + 1) << 5, q * colstride + colb, lane);
#pragma unroll
    for (int q = 0; q < Q; ++q) acc[q] = wmma_bf16(a0, b0[q], acc[q]);

    a0 = load_frag_a(asrc, lda, (ti + 2) << 5, lane);
#pragma unroll
    for (int q = 0; q < Q; ++q)
      b0[q] = load_frag_b(wt, ldb, (ti + 2) << 5, q * colstride + colb, lane);
#pragma unroll
    for (int q = 0; q < Q; ++q) acc[q] = wmma_bf16(a1, b1[q], acc[q]);
  }
  // last pair
  a1 = load_frag_a(asrc, lda, (nT - 1) << 5, lane);
#pragma unroll
  for (int q = 0; q < Q; ++q)
    b1[q] = load_frag_b(wt, ldb, (nT - 1) << 5, q * colstride + colb, lane);
#pragma unroll
  for (int q = 0; q < Q; ++q) acc[q] = wmma_bf16(a0, b0[q], acc[q]);
#pragma unroll
  for (int q = 0; q < Q; ++q) acc[q] = wmma_bf16(a1, b1[q], acc[q]);
}

// ---------------- setup kernels ----------------

// src: R x C fp32 row-major  ->  dst: C x R bf16 row-major (dst[k][n] = src[n][k])
__global__ void conv_tr_k(const float* __restrict__ src, unsigned short* __restrict__ dst,
                          int R, int C) {
  int idx = blockIdx.x * blockDim.x + threadIdx.x;
  if (idx < R * C) {
    int n = idx / C;
    int k = idx - n * C;
    dst[(size_t)k * R + n] = f2bf(src[idx]);
  }
}

__global__ void zero_k(unsigned int* __restrict__ p, int nwords) {
  int i = blockIdx.x * blockDim.x + threadIdx.x;
  if (i < nwords) p[i] = 0u;
}

// ---------------- per-step kernels ----------------

// x = [ emb[tok] , att_ctx ]  -> bf16, 16 x 1024
__global__ void build_x_k(const float* __restrict__ emb, const float* __restrict__ actx,
                          const int* __restrict__ tgt, unsigned short* __restrict__ xbf,
                          int tstep) {
  int idx = blockIdx.x * blockDim.x + threadIdx.x;
  if (idx >= N_ * KX0) return;
  int n = idx >> 10;
  int j = idx & 1023;
  float v;
  if (j < H_) {
    int tok = tgt[n * To_ + tstep];
    v = emb[(size_t)tok * H_ + j];
  } else {
    v = actx[n * D_ + (j - H_)];
  }
  xbf[idx] = f2bf(v);
}

// Fused LSTM layer: one wave per 16 hidden columns computes i/f/g/o gate tiles
// via WMMA (fp32 accum), two ping-pong pipelined K-phases (x then h), then the
// cell update entirely in registers.
__global__ void __launch_bounds__(32) lstm_layer_k(
    const unsigned short* __restrict__ xbf, int Kx,
    const unsigned short* __restrict__ hbf_in,
    const unsigned short* __restrict__ Wiht,   // Kx  x 2048 (K-major)
    const unsigned short* __restrict__ Whht,   // 512 x 2048 (K-major)
    const float* __restrict__ bih, const float* __restrict__ bhh,
    float* __restrict__ c_st, float* __restrict__ h32,
    unsigned short* __restrict__ hbf_out) {
  const int lane = threadIdx.x;
  const int colb = blockIdx.x * 16;
  const int n    = lane & 15;

  v8f acc[4];
#pragma unroll
  for (int q = 0; q < 4; ++q) {
    float b = bih[q * H_ + colb + n] + bhh[q * H_ + colb + n];
#pragma unroll
    for (int r = 0; r < 8; ++r) acc[q][r] = b;
  }

  gemm_pipe<4>(xbf,    Kx, Wiht, G4H, Kx >> 5, colb, H_, lane, acc);  // x phase
  gemm_pipe<4>(hbf_in, H_, Whht, G4H, H_ >> 5, colb, H_, lane, acc);  // h phase

#pragma unroll
  for (int r = 0; r < 8; ++r) {
    int m   = ((lane >> 4) << 3) + r;
    int idx = m * H_ + colb + n;
    float ig = sigf(acc[0][r]);
    float fg = sigf(acc[1][r]);
    float gg = tanhf(acc[2][r]);
    float og = sigf(acc[3][r]);
    float c  = fg * c_st[idx] + ig * gg;
    c_st[idx] = c;
    float h   = og * tanhf(c);
    h32[idx]  = h;
    hbf_out[idx] = f2bf(h);
  }
}

// Fused attention: scores -> masked softmax -> ali (to output) -> context,
// also builds the proj input concat [dec_out, ctx] in bf16.
// Decoder row staged in LDS, dot products in float4.
__global__ void __launch_bounds__(256) attn_k(
    const float* __restrict__ enc, const int* __restrict__ enc_len,
    const float* __restrict__ dec,           // h32 layer 2, 16 x 512
    float* __restrict__ alis,                // N x To x Ti
    float* __restrict__ actx,                // 16 x 512 (carry)
    unsigned short* __restrict__ xcat,       // 16 x 1024 bf16
    int tstep) {
  __shared__ float4 dn4[D_ / 4];             // 2 KB: decoder row
  __shared__ float sc[Ti_];
  __shared__ float red[256];
  const int n   = blockIdx.x;
  const int tid = threadIdx.x;
  int elen = enc_len[n];
  if (elen < 1) elen = 1;
  const float scale = 0.044194173824159216f;  // 1/sqrt(512)
  const float* dn = dec + n * D_;

  for (int i = tid; i < D_ / 4; i += 256)
    dn4[i] = reinterpret_cast<const float4*>(dn)[i];
  __syncthreads();

  for (int t = tid; t < Ti_; t += 256) {
    const float4* e4 = reinterpret_cast<const float4*>(enc + ((size_t)n * Ti_ + t) * D_);
    float s = 0.f;
    for (int d = 0; d < D_ / 4; ++d) {
      float4 ev = e4[d], dv = dn4[d];
      s += ev.x * dv.x + ev.y * dv.y + ev.z * dv.z + ev.w * dv.w;
    }
    sc[t] = (t < elen) ? s * scale : -1e30f;
  }
  __syncthreads();

  float lm = -3.0e38f;
  for (int t = tid; t < Ti_; t += 256) lm = fmaxf(lm, sc[t]);
  red[tid] = lm;
  __syncthreads();
  for (int s = 128; s > 0; s >>= 1) {
    if (tid < s) red[tid] = fmaxf(red[tid], red[tid + s]);
    __syncthreads();
  }
  float mx = red[0];
  __syncthreads();

  float ls = 0.f;
  for (int t = tid; t < Ti_; t += 256) {
    float e = __expf(sc[t] - mx);
    sc[t] = e;
    ls += e;
  }
  red[tid] = ls;
  __syncthreads();
  for (int s = 128; s > 0; s >>= 1) {
    if (tid < s) red[tid] += red[tid + s];
    __syncthreads();
  }
  float inv = 1.0f / red[0];
  __syncthreads();

  float* arow = alis + ((size_t)n * To_ + tstep) * Ti_;
  for (int t = tid; t < Ti_; t += 256) arow[t] = sc[t] * inv;

  for (int d = tid; d < D_; d += 256) {
    float a = 0.f;
    for (int t = 0; t < elen; ++t) a += sc[t] * enc[((size_t)n * Ti_ + t) * D_ + d];
    a *= inv;
    actx[n * D_ + d] = a;
    xcat[n * KX0 + H_ + d] = f2bf(a);      // second half: ctx
    xcat[n * KX0 + d]      = f2bf(dn[d]);  // first half: dec_out
  }
}

// p = relu([dec_out, ctx] @ Wp.T + bp) -> bf16 16x512 (ping-pong pipelined)
__global__ void __launch_bounds__(32) proj_k(
    const unsigned short* __restrict__ xcat,   // 16 x 1024
    const unsigned short* __restrict__ Wpt,    // 1024 x 512 (K-major)
    const float* __restrict__ bp,
    unsigned short* __restrict__ pbf) {
  const int lane = threadIdx.x;
  const int colb = blockIdx.x * 16;
  const int n    = lane & 15;
  float b = bp[colb + n];
  v8f acc;
#pragma unroll
  for (int r = 0; r < 8; ++r) acc[r] = b;

  gemm_pipe<1>(xcat, KX0, Wpt, D_, KX0 >> 5, colb, 0, lane, &acc);

#pragma unroll
  for (int r = 0; r < 8; ++r) {
    int m = ((lane >> 4) << 3) + r;
    pbf[m * D_ + colb + n] = f2bf(fmaxf(acc[r], 0.0f));
  }
}

// pred = p @ Wv.T + bv -> fp32 directly into outs[:, tstep, :] (ping-pong pipelined)
__global__ void __launch_bounds__(32) pred_k(
    const unsigned short* __restrict__ pbf,   // 16 x 512
    const unsigned short* __restrict__ Wvt,   // 512 x 8000 (K-major)
    const float* __restrict__ bv,
    float* __restrict__ outs, int tstep) {
  const int lane = threadIdx.x;
  const int colb = blockIdx.x * 16;
  const int n    = lane & 15;
  float b = bv[colb + n];
  v8f acc;
#pragma unroll
  for (int r = 0; r < 8; ++r) acc[r] = b;

  gemm_pipe<1>(pbf, D_, Wvt, V_, D_ >> 5, colb, 0, lane, &acc);

#pragma unroll
  for (int r = 0; r < 8; ++r) {
    int m = ((lane >> 4) << 3) + r;
    outs[((size_t)m * To_ + tstep) * V_ + colb + n] = acc[r];
  }
}

// ---------------- host driver ----------------

extern "C" void kernel_launch(void* const* d_in, const int* in_sizes, int n_in,
                              void* d_out, int out_size, void* d_ws, size_t ws_size,
                              hipStream_t stream) {
  (void)in_sizes; (void)n_in; (void)out_size; (void)ws_size;

  const float* enc   = (const float*)d_in[0];
  const float* emb   = (const float*)d_in[1];
  const float* Wih0  = (const float*)d_in[2];
  const float* Whh0  = (const float*)d_in[3];
  const float* Wih12 = (const float*)d_in[4];
  const float* Whh12 = (const float*)d_in[5];
  const float* bih   = (const float*)d_in[6];
  const float* bhh   = (const float*)d_in[7];
  const float* Wp    = (const float*)d_in[8];
  const float* bp    = (const float*)d_in[9];
  const float* Wv    = (const float*)d_in[10];
  const float* bv    = (const float*)d_in[11];
  const int* enc_len = (const int*)d_in[12];
  const int* tgt     = (const int*)d_in[13];

  float* outs = (float*)d_out;
  float* alis = outs + (size_t)N_ * To_ * V_;

  char* ws = (char*)d_ws;
  size_t off = 0;
  auto alloc = [&](size_t bytes) -> char* {
    char* p = ws + off;
    off += (bytes + 255) & ~(size_t)255;
    return p;
  };

  unsigned short* Wih0t = (unsigned short*)alloc((size_t)KX0 * G4H * 2);
  unsigned short* Whh0t = (unsigned short*)alloc((size_t)H_  * G4H * 2);
  unsigned short* Wih1t = (unsigned short*)alloc((size_t)H_  * G4H * 2);
  unsigned short* Wih2t = (unsigned short*)alloc((size_t)H_  * G4H * 2);
  unsigned short* Whh1t = (unsigned short*)alloc((size_t)H_  * G4H * 2);
  unsigned short* Whh2t = (unsigned short*)alloc((size_t)H_  * G4H * 2);
  unsigned short* Wpt   = (unsigned short*)alloc((size_t)KX0 * D_  * 2);
  unsigned short* Wvt   = (unsigned short*)alloc((size_t)D_  * V_  * 2);
  unsigned short* xbf   = (unsigned short*)alloc((size_t)N_ * KX0 * 2);
  unsigned short* pbf   = (unsigned short*)alloc((size_t)N_ * D_  * 2);
  // state block (zeroed every call): hbufA, hbufB, h32, c32, actx — contiguous
  unsigned short* hbufA = (unsigned short*)alloc((size_t)3 * N_ * H_ * 2);
  unsigned short* hbufB = (unsigned short*)alloc((size_t)3 * N_ * H_ * 2);
  float* h32  = (float*)alloc((size_t)3 * N_ * H_ * 4);
  float* c32  = (float*)alloc((size_t)3 * N_ * H_ * 4);
  float* actx = (float*)alloc((size_t)N_ * D_ * 4);

  // zero carries (hbufA..actx are contiguous, all 256B-padded)
  {
    size_t bytes = (size_t)((char*)actx - (char*)hbufA) +
                   (((size_t)N_ * D_ * 4 + 255) & ~(size_t)255);
    int words = (int)(bytes / 4);
    zero_k<<<(words + 255) / 256, 256, 0, stream>>>((unsigned int*)hbufA, words);
  }

  // weight conversion + transpose to K-major bf16 (recomputed every call: deterministic)
  auto tr = [&](const float* s, unsigned short* d, int R, int C) {
    int total = R * C;
    conv_tr_k<<<(total + 255) / 256, 256, 0, stream>>>(s, d, R, C);
  };
  tr(Wih0,                     Wih0t, G4H, KX0);
  tr(Whh0,                     Whh0t, G4H, H_);
  tr(Wih12,                    Wih1t, G4H, H_);
  tr(Wih12 + (size_t)G4H * H_, Wih2t, G4H, H_);
  tr(Whh12,                    Whh1t, G4H, H_);
  tr(Whh12 + (size_t)G4H * H_, Whh2t, G4H, H_);
  tr(Wp,                       Wpt,   D_,  KX0);
  tr(Wv,                       Wvt,   V_,  D_);

  const int HS = N_ * H_;  // per-layer state slice (elements)

  for (int t = 0; t < To_; ++t) {
    const int par = t & 1;
    unsigned short* hin  = par ? hbufB : hbufA;
    unsigned short* hout = par ? hbufA : hbufB;

    build_x_k<<<(N_ * KX0 + 255) / 256, 256, 0, stream>>>(emb, actx, tgt, xbf, t);

    lstm_layer_k<<<H_ / 16, 32, 0, stream>>>(xbf, KX0, hin, Wih0t, Whh0t,
                                             bih, bhh, c32, h32, hout);
    lstm_layer_k<<<H_ / 16, 32, 0, stream>>>(hout, H_, hin + HS, Wih1t, Whh1t,
                                             bih + G4H, bhh + G4H,
                                             c32 + HS, h32 + HS, hout + HS);
    lstm_layer_k<<<H_ / 16, 32, 0, stream>>>(hout + HS, H_, hin + 2 * HS, Wih2t, Whh2t,
                                             bih + 2 * G4H, bhh + 2 * G4H,
                                             c32 + 2 * HS, h32 + 2 * HS, hout + 2 * HS);

    attn_k<<<N_, 256, 0, stream>>>(enc, enc_len, h32 + 2 * HS, alis, actx, xbf, t);

    proj_k<<<D_ / 16, 32, 0, stream>>>(xbf, Wpt, bp, pbf);

    pred_k<<<V_ / 16, 32, 0, stream>>>(pbf, Wvt, bv, outs, t);
  }
}
